// QLSTM_65481071406534
// MI455X (gfx1250) — compile-verified, matrix operand order checked by reading, not run on hardware
//
#include <hip/hip_runtime.h>

typedef __attribute__((ext_vector_type(16))) _Float16 v16h;
typedef __attribute__((ext_vector_type(8)))  float    v8f;
typedef __attribute__((ext_vector_type(4)))  float    v4f;

#define T_STEPS 256
#define BATCH   2048
#define DIN     128
#define NQ      16
#define NCOL    64          // 4 gates * NQ
#define WROW    (DIN + NQ)  // 144
#define TC      64          // time chunk

// ---------------------------------------------------------------------------
// Phase 1: z_x[t,b,g,n] = sum_k x[t,b,k] * W[g,n,k]  + (b[g,n] + theta[g,n])
// One wave computes a 16-row x 64-col tile via 16 x v_wmma_f32_16x16x32_f16.
// x is streamed with NT hints so the 192MB L2 retains the zx chunk instead.
// ---------------------------------------------------------------------------
__global__ __launch_bounds__(256)
void qlstm_phase1_wmma(const float* __restrict__ x,    // [TC*B, 128] (chunk)
                       const float* __restrict__ W,    // [4,16,144]
                       const float* __restrict__ bb,   // [4,16]
                       const float* __restrict__ th,   // [4,16]
                       float* __restrict__ zx)         // [TC*B, 64]
{
    __shared__ _Float16 wlds[NCOL * DIN];   // 16 KB: W[:, :, :128] as f16

    // cooperative f32->f16 weight stage (columns 0..127 of each W row)
    for (int i = threadIdx.x; i < NCOL * DIN; i += 256) {
        int r = i >> 7, k = i & 127;
        wlds[i] = (_Float16)W[r * WROW + k];
    }
    __syncthreads();

    const int lane   = threadIdx.x & 31;
    const int lane16 = lane & 15;
    const int laneHi = lane >> 4;          // which half-wave
    const int waveId = blockIdx.x * (blockDim.x >> 5) + (threadIdx.x >> 5);
    const int nWaves = gridDim.x * (blockDim.x >> 5);

    // B-matrix fragments: bf[gate][kstep], kept in registers for all tiles.
    // 16-bit matrix layout: lanes 0-15 -> K {kb..kb+7, kb+16..kb+23},
    //                       lanes 16-31 -> K {kb+8..kb+15, kb+24..kb+31}.
    v16h bf[4][4];
    #pragma unroll
    for (int g = 0; g < 4; ++g) {
        const _Float16* wrow = &wlds[(g * 16 + lane16) * DIN];
        #pragma unroll
        for (int ks = 0; ks < 4; ++ks) {
            const int kb = ks * 32 + laneHi * 8;
            #pragma unroll
            for (int j = 0; j < 8; ++j) {
                bf[g][ks][j]     = wrow[kb + j];
                bf[g][ks][j + 8] = wrow[kb + 16 + j];
            }
        }
    }
    // fused bias: b + theta (column = g*16 + lane16)
    float bias[4];
    #pragma unroll
    for (int g = 0; g < 4; ++g)
        bias[g] = bb[g * 16 + lane16] + th[g * 16 + lane16];

    const int nTiles = (TC * BATCH) / 16;
    for (int tile = waveId; tile < nTiles; tile += nWaves) {
        const int r0 = tile * 16;
        const float* arow = x + (size_t)(r0 + lane16) * DIN;

        v8f acc[4] = {};
        #pragma unroll
        for (int ks = 0; ks < 4; ++ks) {
            const int kb = ks * 32 + laneHi * 8;
            // stream-once A tile: nontemporal b128 loads
            const v4f p0 = __builtin_nontemporal_load((const v4f*)(arow + kb));
            const v4f p1 = __builtin_nontemporal_load((const v4f*)(arow + kb + 4));
            const v4f p2 = __builtin_nontemporal_load((const v4f*)(arow + kb + 16));
            const v4f p3 = __builtin_nontemporal_load((const v4f*)(arow + kb + 20));
            v16h af;
            #pragma unroll
            for (int j = 0; j < 4; ++j) {
                af[j]      = (_Float16)p0[j];
                af[j + 4]  = (_Float16)p1[j];
                af[j + 8]  = (_Float16)p2[j];
                af[j + 12] = (_Float16)p3[j];
            }
            #pragma unroll
            for (int g = 0; g < 4; ++g)
                acc[g] = __builtin_amdgcn_wmma_f32_16x16x32_f16(
                    false, af, false, bf[g][ks], (short)0, acc[g], false, false);
        }
        // D layout: VGPR r -> (M = r + 8*laneHi, N = lane16)
        // zx stores stay regular-temporal: phase 2 re-reads them from L2.
        #pragma unroll
        for (int g = 0; g < 4; ++g) {
            #pragma unroll
            for (int r = 0; r < 8; ++r) {
                const int m = r + laneHi * 8;
                zx[(size_t)(r0 + m) * NCOL + g * 16 + lane16] = acc[g][r] + bias[g];
            }
        }
    }
}

// ---------------------------------------------------------------------------
// Phase 2: per-batch-row independent LSTM scan over a time chunk.
// 16 lanes per row (lane = qubit n); each lane carries h[n], c[n].
// ---------------------------------------------------------------------------
__device__ __forceinline__ float fast_sigmoid(float v) {
    return 1.0f / (1.0f + __expf(-v));
}
__device__ __forceinline__ float fast_tanh(float v) {
    return 1.0f - 2.0f / (1.0f + __expf(2.0f * v));
}

__global__ __launch_bounds__(256)
void qlstm_phase2_scan(const float* __restrict__ zx,   // [TC, B, 4, 16]
                       const float* __restrict__ W,    // [4,16,144]
                       float* __restrict__ out,        // outputs|hx|cx
                       float* __restrict__ hstate,     // [B,16] carry
                       float* __restrict__ cstate,     // [B,16] carry
                       int t0, int first, int last)
{
    const int n   = threadIdx.x & 15;
    const int row = blockIdx.x * 16 + (threadIdx.x >> 4);

    // recurrent weights: Wh[g][k] = W[g, n, 128+k]
    float wh[4][NQ];
    #pragma unroll
    for (int g = 0; g < 4; ++g) {
        const float* wr = W + (size_t)(g * 16 + n) * WROW + DIN;
        #pragma unroll
        for (int k = 0; k < NQ; ++k) wh[g][k] = wr[k];
    }

    float h = 0.0f, c = 0.0f;
    if (!first) {
        h = hstate[(size_t)row * NQ + n];
        c = cstate[(size_t)row * NQ + n];
    }

    for (int lt = 0; lt < TC; ++lt) {
        const float* zp = zx + ((size_t)lt * BATCH + row) * NCOL;
        // zx is read exactly once (producer-consumer through L2) -> NT
        float a0 = __builtin_nontemporal_load(zp + n);
        float a1 = __builtin_nontemporal_load(zp + 16 + n);
        float a2 = __builtin_nontemporal_load(zp + 32 + n);
        float a3 = __builtin_nontemporal_load(zp + 48 + n);

        // + h . Wh  (broadcast h[k] across the 16-lane row group)
        #pragma unroll
        for (int k = 0; k < NQ; ++k) {
            const float hk = __shfl(h, k, 16);
            a0 = fmaf(hk, wh[0][k], a0);
            a1 = fmaf(hk, wh[1][k], a1);
            a2 = fmaf(hk, wh[2][k], a2);
            a3 = fmaf(hk, wh[3][k], a3);
        }

        float q0 = __cosf(a0), q1 = __cosf(a1), q2 = __cosf(a2), q3 = __cosf(a3);

        // inclusive prefix product over n (CNOT-chain cumprod), 4 shuffle stages
        #pragma unroll
        for (int off = 1; off < 16; off <<= 1) {
            const float p0 = __shfl_up(q0, off, 16);
            const float p1 = __shfl_up(q1, off, 16);
            const float p2 = __shfl_up(q2, off, 16);
            const float p3 = __shfl_up(q3, off, 16);
            if (n >= off) { q0 *= p0; q1 *= p1; q2 *= p2; q3 *= p3; }
        }

        const float f  = fast_sigmoid(q0);
        const float ii = fast_sigmoid(q1);
        const float gg = fast_tanh(q2);
        const float oo = fast_sigmoid(q3);
        c = f * c + ii * gg;
        h = oo * fast_tanh(c);

        // outputs are never re-read on device -> NT store
        __builtin_nontemporal_store(h, &out[((size_t)(t0 + lt) * BATCH + row) * NQ + n]);
    }

    hstate[(size_t)row * NQ + n] = h;
    cstate[(size_t)row * NQ + n] = c;
    if (last) {
        const size_t tail = (size_t)T_STEPS * BATCH * NQ;
        out[tail + (size_t)row * NQ + n]                      = h;  // hx
        out[tail + (size_t)BATCH * NQ + (size_t)row * NQ + n] = c;  // cx
    }
}

// ---------------------------------------------------------------------------
extern "C" void kernel_launch(void* const* d_in, const int* in_sizes, int n_in,
                              void* d_out, int out_size, void* d_ws, size_t ws_size,
                              hipStream_t stream) {
    const float* x  = (const float*)d_in[0];   // [T,B,128]
    const float* W  = (const float*)d_in[1];   // [4,16,144]
    const float* bb = (const float*)d_in[2];   // [4,16]
    const float* th = (const float*)d_in[3];   // [4,16]
    float* out = (float*)d_out;

    float* zxbuf  = (float*)d_ws;                                  // 32 MB chunk
    float* hstate = zxbuf + (size_t)TC * BATCH * NCOL;
    float* cstate = hstate + (size_t)BATCH * NQ;

    for (int t0 = 0; t0 < T_STEPS; t0 += TC) {
        qlstm_phase1_wmma<<<512, 256, 0, stream>>>(
            x + (size_t)t0 * BATCH * DIN, W, bb, th, zxbuf);
        qlstm_phase2_scan<<<BATCH / 16, 256, 0, stream>>>(
            zxbuf, W, out, hstate, cstate,
            t0, t0 == 0 ? 1 : 0, (t0 + TC >= T_STEPS) ? 1 : 0);
    }
}
